// WassersteinWrapper_26139170963916
// MI455X (gfx1250) — matrix operand order repaired and checked by reading, not compiled, not run on hardware
//
#include <hip/hip_runtime.h>
#include <hip/hip_bf16.h>
#include <math.h>

#define N_PTS   8192
#define DIMS    64
#define BLURF   0.01f
#define BLOGC   (-9.010913347279288f)   /* -log(8192) */
#define MAX_ITERS 40
#define JSPLIT  8                        /* column-tile chunks (grid.y) */
#define PLANE   (JSPLIT * N_PTS)         /* one (m or s) plane per softmin slot */
#define SLOTSZ  (2 * PLANE)              /* m-plane + s-plane */

typedef __attribute__((ext_vector_type(16))) __bf16 v16bf;
typedef __attribute__((ext_vector_type(8)))  __bf16 v8bf;
typedef __attribute__((ext_vector_type(8)))  float  v8f;

__device__ __forceinline__ unsigned short f2bf(float x) {
    unsigned int u = __float_as_uint(x);
    unsigned int r = u + 0x7FFFu + ((u >> 16) & 1u);   // round-to-nearest-even
    return (unsigned short)(r >> 16);
}

// ---------------------------------------------------------------------------
// Row norms + bf16 conversion for x and y (16384 rows total).
// ---------------------------------------------------------------------------
__global__ void prep_kernel(const float* __restrict__ x, const float* __restrict__ y,
                            float* __restrict__ x2, float* __restrict__ y2,
                            unsigned short* __restrict__ xb, unsigned short* __restrict__ yb) {
    int row = blockIdx.x * blockDim.x + threadIdx.x;   // 0..16383
    const float* src; float* sq; unsigned short* dst; int r;
    if (row < N_PTS) { src = x; sq = x2; dst = xb; r = row; }
    else             { src = y; sq = y2; dst = yb; r = row - N_PTS; }
    float acc = 0.f;
    for (int k = 0; k < DIMS; ++k) {
        float v = src[(size_t)r * DIMS + k];
        acc += v * v;
        dst[(size_t)r * DIMS + k] = f2bf(v);
    }
    sq[r] = acc;
}

// ---------------------------------------------------------------------------
// Per-dimension min/max over the union of x and y (one block per dim).
// ---------------------------------------------------------------------------
__global__ void minmax_kernel(const float* __restrict__ x, const float* __restrict__ y,
                              float* __restrict__ mins, float* __restrict__ maxs) {
    int d = blockIdx.x, t = threadIdx.x;
    float mn = 3.4e38f, mx = -3.4e38f;
    for (int r = t; r < N_PTS; r += blockDim.x) {
        float a = x[(size_t)r * DIMS + d];
        float b = y[(size_t)r * DIMS + d];
        mn = fminf(mn, fminf(a, b));
        mx = fmaxf(mx, fmaxf(a, b));
    }
    __shared__ float smn[256], smx[256];
    smn[t] = mn; smx[t] = mx; __syncthreads();
    for (int s = 128; s > 0; s >>= 1) {
        if (t < s) { smn[t] = fminf(smn[t], smn[t + s]); smx[t] = fmaxf(smx[t], smx[t + s]); }
        __syncthreads();
    }
    if (t == 0) { mins[d] = smn[0]; maxs[d] = smx[0]; }
}

// ---------------------------------------------------------------------------
// Epsilon schedule: [diam] + geometric(diam -> BLUR, ratio 0.5) + [BLUR].
// ---------------------------------------------------------------------------
__global__ void sched_kernel(const float* __restrict__ mins, const float* __restrict__ maxs,
                             float* __restrict__ eps, int* __restrict__ neps) {
    if (threadIdx.x != 0) return;
    float s = 0.f;
    for (int d = 0; d < DIMS; ++d) { float v = maxs[d] - mins[d]; s += v * v; }
    float diam = __builtin_amdgcn_sqrtf(s);
    int n = 0;
    eps[n++] = diam;                     // eps0 (also first loop eps via mids[0]==diam)
    float e = diam;
    while (e > BLURF && n < 58) { eps[n++] = e; e *= 0.5f; }
    eps[n++] = BLURF;
    eps[62]  = BLURF;                    // dedicated slot for the final extrapolation
    *neps = n;
}

// ---------------------------------------------------------------------------
// Fused softmin partial with online (branchless, 1-exp) logsumexp:
//   t_ij = blog + gscale*g[j]/eps - C_ij/eps,
//   C_ij = sqrt(max(|x_i|^2 + |y_j|^2 - 2 x_i.y_j, 0))  via v_wmma bf16.
// One wave owns a 16-row strip; grid.y splits the 512 column tiles into JSPLIT
// chunks; each chunk writes its private (m, s) pair per row (no atomics).
// ---------------------------------------------------------------------------
__global__ __launch_bounds__(256) void softmin_partial(
    const unsigned short* __restrict__ Arows,   // bf16 rows (i index)
    const unsigned short* __restrict__ Brows,   // bf16 rows (j index)
    const float* __restrict__ rowsq,
    const float* __restrict__ colsq,
    const float* __restrict__ gvec, float gscale,
    const float* __restrict__ epsp,
    float* __restrict__ mout, float* __restrict__ sout,   // JSPLIT*N_PTS planes
    const int* __restrict__ neps, int iter)
{
    if (iter >= 0 && iter >= *neps) return;      // uniform: whole wave exits

    const float eps     = epsp[0];
    const float inv_eps = 1.0f / eps;

    const int lane = threadIdx.x & 31;
    const int wave = threadIdx.x >> 5;
    const int i0   = (blockIdx.x * 8 + wave) * 16;
    const int half = lane >> 4;                  // K-half for A/B per ISA layout
    const int ml   = lane & 15;

    // A tile (16x64 bf16): row = i0+ml; lane holds K in {0..15}+16*half per 32-chunk
    const unsigned short* abase = Arows + (size_t)(i0 + ml) * DIMS;
    v8bf a0lo = *reinterpret_cast<const v8bf*>(abase + 0  + half * 16);
    v8bf a0hi = *reinterpret_cast<const v8bf*>(abase + 8  + half * 16);
    v8bf a1lo = *reinterpret_cast<const v8bf*>(abase + 32 + half * 16);
    v8bf a1hi = *reinterpret_cast<const v8bf*>(abase + 40 + half * 16);
    v16bf A0, A1;
    #pragma unroll
    for (int e = 0; e < 8; ++e) {
        A0[e] = a0lo[e]; A0[8 + e] = a0hi[e];
        A1[e] = a1lo[e]; A1[8 + e] = a1hi[e];
    }

    float rs[8];
    #pragma unroll
    for (int r = 0; r < 8; ++r) rs[r] = rowsq[i0 + r + 8 * half];   // C/D layout rows

    float sAcc[8], mAcc[8];
    #pragma unroll
    for (int r = 0; r < 8; ++r) { sAcc[r] = 0.f; mAcc[r] = -3.4e38f; }

    const int tiles = N_PTS / 16 / JSPLIT;       // 64
    const int jt0 = blockIdx.y * tiles, jt1 = jt0 + tiles;

    #pragma unroll 2
    for (int jt = jt0; jt < jt1; ++jt) {
        int col = jt * 16 + ml;
        const unsigned short* bbase = Brows + (size_t)col * DIMS + half * 16;
        __builtin_prefetch(bbase + 1024, 0, 1);                     // next B tile
        v16bf B0 = *reinterpret_cast<const v16bf*>(bbase);          // K 0..31 chunk
        v16bf B1 = *reinterpret_cast<const v16bf*>(bbase + 32);     // K 32..63 chunk

        v8f c = {};
        c = __builtin_amdgcn_wmma_f32_16x16x32_bf16(false, A0, false, B0, (short)0, c, false, false);
        c = __builtin_amdgcn_wmma_f32_16x16x32_bf16(false, A1, false, B1, (short)0, c, false, false);

        float cq = colsq[col];
        float hj = BLOGC + gscale * gvec[col] * inv_eps;            // branchless
        #pragma unroll
        for (int r = 0; r < 8; ++r) {
            float d2 = fmaxf(rs[r] + cq - 2.0f * c[r], 0.f);
            float cd = __builtin_amdgcn_sqrtf(d2);                  // raw v_sqrt_f32
            float t  = hj - cd * inv_eps;
            // branchless online LSE: one exp per element
            float d  = t - mAcc[r];
            float e  = __expf(-fabsf(d));
            bool  up = d > 0.f;
            sAcc[r]  = up ? fmaf(sAcc[r], e, 1.f) : (sAcc[r] + e);
            mAcc[r]  = fmaxf(mAcc[r], t);
        }
    }

    // Combine (m,s) pairs across the 16 lanes that share the same row set.
    #pragma unroll
    for (int r = 0; r < 8; ++r) {
        float m = mAcc[r], s = sAcc[r];
        for (int msk = 8; msk >= 1; msk >>= 1) {
            float mo = __shfl_xor(m, msk, 16);
            float so = __shfl_xor(s, msk, 16);
            float M  = fmaxf(m, mo);
            s = s * __expf(m - M) + so * __expf(mo - M);
            m = M;
        }
        mAcc[r] = m; sAcc[r] = s;
    }
    if (ml == 0) {
        int base = blockIdx.y * N_PTS;
        #pragma unroll
        for (int r = 0; r < 8; ++r) {
            int i = i0 + r + 8 * half;
            mout[base + i] = mAcc[r];
            sout[base + i] = sAcc[r];
        }
    }
}

// ---------------------------------------------------------------------------
// Merge JSPLIT (m,s) pairs per row per slot, form ft = -eps*lse, and either
// overwrite (init, mode==0) or average (loop, mode==1) into the duals.
// ---------------------------------------------------------------------------
__device__ __forceinline__ float merge_lse(const float* __restrict__ sb, int slot, int i,
                                           float eps) {
    const float* mp = sb + (size_t)slot * SLOTSZ;
    const float* sp = mp + PLANE;
    float M = -3.4e38f;
    #pragma unroll
    for (int c = 0; c < JSPLIT; ++c) M = fmaxf(M, mp[c * N_PTS + i]);
    float S = 0.f;
    #pragma unroll
    for (int c = 0; c < JSPLIT; ++c)
        S += sp[c * N_PTS + i] * __expf(mp[c * N_PTS + i] - M);
    return -eps * (M + __logf(S));
}

__global__ __launch_bounds__(1024) void finalize_update(
    float* __restrict__ fba, float* __restrict__ gab,
    float* __restrict__ faa, float* __restrict__ gbb,
    const float* __restrict__ sb, const float* __restrict__ epsp,
    const int* __restrict__ neps, int iter, int mode)
{
    if (iter >= 0 && iter >= *neps) return;
    float eps = epsp[0];
    int t = threadIdx.x + blockIdx.x * blockDim.x;
    for (int i = t; i < N_PTS; i += gridDim.x * blockDim.x) {
        float ft0 = merge_lse(sb, 0, i, eps);   // ft_ba
        float ft1 = merge_lse(sb, 1, i, eps);   // gt_ab
        float ft2 = merge_lse(sb, 2, i, eps);   // ft_aa
        float ft3 = merge_lse(sb, 3, i, eps);   // gt_bb
        fba[i] = mode ? 0.5f * (fba[i] + ft0) : ft0;
        gab[i] = mode ? 0.5f * (gab[i] + ft1) : ft1;
        faa[i] = mode ? 0.5f * (faa[i] + ft2) : ft2;
        gbb[i] = mode ? 0.5f * (gbb[i] + ft3) : ft3;
    }
}

// ---------------------------------------------------------------------------
// Final extrapolation (eps = BLUR) + divergence reduction:
//   S = (1/N) sum(f_ba - f_aa) + (1/N) sum(g_ab - g_bb)
// ---------------------------------------------------------------------------
__global__ __launch_bounds__(1024) void final_reduce(
    const float* __restrict__ sb, float* __restrict__ out)
{
    const float eps = BLURF;
    int t = threadIdx.x;
    float acc = 0.f;
    for (int i = t; i < N_PTS; i += 1024) {
        float ft_ba = merge_lse(sb, 0, i, eps);
        float gt_ab = merge_lse(sb, 1, i, eps);
        float ft_aa = merge_lse(sb, 2, i, eps);
        float gt_bb = merge_lse(sb, 3, i, eps);
        acc += (ft_ba - ft_aa) + (gt_ab - gt_bb);
    }
    __shared__ float red[1024];
    red[t] = acc; __syncthreads();
    for (int st = 512; st > 0; st >>= 1) {
        if (t < st) red[t] += red[t + st];
        __syncthreads();
    }
    if (t == 0) out[0] = red[0] * (1.0f / N_PTS);
}

// ---------------------------------------------------------------------------
extern "C" void kernel_launch(void* const* d_in, const int* in_sizes, int n_in,
                              void* d_out, int out_size, void* d_ws, size_t ws_size,
                              hipStream_t stream) {
    const float* x = (const float*)d_in[0];
    const float* y = (const float*)d_in[1];

    float* f    = (float*)d_ws;
    float* x2   = f;                         // 8192
    float* y2   = f + 8192;                  // 8192
    float* fba  = f + 16384;                 // duals
    float* gab  = f + 24576;
    float* faa  = f + 32768;
    float* gbb  = f + 40960;
    float* sb   = f + 49152;                 // 4 slots * (m+s planes) = 524288 floats
    float* mins = f + 573440;                // 64
    float* maxs = f + 573504;                // 64
    float* eps  = f + 573568;                // 64-entry schedule (+slot 62 = BLUR)
    int*   neps = (int*)(f + 573632);
    unsigned short* xb = (unsigned short*)(f + 573696);  // 32-byte aligned
    unsigned short* yb = xb + (size_t)N_PTS * DIMS;

    // slot plane pointers (m-plane; s-plane is +PLANE inside the kernel helpers)
    float* m0 = sb + 0 * SLOTSZ, *s0 = m0 + PLANE;
    float* m1 = sb + 1 * SLOTSZ, *s1 = m1 + PLANE;
    float* m2 = sb + 2 * SLOTSZ, *s2 = m2 + PLANE;
    float* m3 = sb + 3 * SLOTSZ, *s3 = m3 + PLANE;

    dim3 pgrid(64, JSPLIT);

    prep_kernel  <<<64, 256, 0, stream>>>(x, y, x2, y2, xb, yb);
    minmax_kernel<<<64, 256, 0, stream>>>(x, y, mins, maxs);
    sched_kernel <<<1, 32, 0, stream>>>(mins, maxs, eps, neps);

    // ---- init softmins at eps0 = diameter (h = blog, gscale = 0) ----
    softmin_partial<<<pgrid, 256, 0, stream>>>(xb, yb, x2, y2, gab, 0.f, eps, m0, s0, neps, -1);
    softmin_partial<<<pgrid, 256, 0, stream>>>(yb, xb, y2, x2, fba, 0.f, eps, m1, s1, neps, -1);
    softmin_partial<<<pgrid, 256, 0, stream>>>(xb, xb, x2, x2, faa, 0.f, eps, m2, s2, neps, -1);
    softmin_partial<<<pgrid, 256, 0, stream>>>(yb, yb, y2, y2, gbb, 0.f, eps, m3, s3, neps, -1);
    finalize_update<<<8, 1024, 0, stream>>>(fba, gab, faa, gbb, sb, eps, neps, -1, 0);

    // ---- annealing loop (fixed launch count, data-dependent early exit) ----
    for (int it = 0; it < MAX_ITERS; ++it) {
        softmin_partial<<<pgrid, 256, 0, stream>>>(xb, yb, x2, y2, gab, 1.f, eps + it, m0, s0, neps, it);
        softmin_partial<<<pgrid, 256, 0, stream>>>(yb, xb, y2, x2, fba, 1.f, eps + it, m1, s1, neps, it);
        softmin_partial<<<pgrid, 256, 0, stream>>>(xb, xb, x2, x2, faa, 1.f, eps + it, m2, s2, neps, it);
        softmin_partial<<<pgrid, 256, 0, stream>>>(yb, yb, y2, y2, gbb, 1.f, eps + it, m3, s3, neps, it);
        finalize_update<<<8, 1024, 0, stream>>>(fba, gab, faa, gbb, sb, eps + it, neps, it, 1);
    }

    // ---- final extrapolation at eps = BLUR (slot 62) + reduction ----
    softmin_partial<<<pgrid, 256, 0, stream>>>(xb, yb, x2, y2, gab, 1.f, eps + 62, m0, s0, neps, -1);
    softmin_partial<<<pgrid, 256, 0, stream>>>(yb, xb, y2, x2, fba, 1.f, eps + 62, m1, s1, neps, -1);
    softmin_partial<<<pgrid, 256, 0, stream>>>(xb, xb, x2, x2, faa, 1.f, eps + 62, m2, s2, neps, -1);
    softmin_partial<<<pgrid, 256, 0, stream>>>(yb, yb, y2, y2, gbb, 1.f, eps + 62, m3, s3, neps, -1);
    final_reduce<<<1, 1024, 0, stream>>>(sb, (float*)d_out);
}
